// GCN_68143951118655
// MI455X (gfx1250) — matrix operand order repaired and checked by reading, not compiled, not run on hardware
//
#include <hip/hip_runtime.h>

typedef __attribute__((ext_vector_type(16))) _Float16 v16h;
typedef __attribute__((ext_vector_type(8)))  float    v8f;
typedef __attribute__((__vector_size__(16))) int      v4i;

#define F_DIM 256   // input features == hidden width
#define C_DIM 64    // head output width

// ---- CDNA5 async global->LDS path (guarded; falls back to sync vectorized copy)
#if defined(__has_builtin)
#  if __has_builtin(__builtin_amdgcn_global_load_async_to_lds_b128)
#    define HAVE_ASYNC_LDS 1
#  endif
#endif
#ifndef HAVE_ASYNC_LDS
#  define HAVE_ASYNC_LDS 0
#endif

#if HAVE_ASYNC_LDS
typedef __attribute__((address_space(1))) v4i* as1_v4i;
typedef __attribute__((address_space(3))) v4i* as3_v4i;
#endif

// ---------------- utility kernels ----------------
__global__ void k_fill(float* p, int n, float v) {
  int i = blockIdx.x * blockDim.x + threadIdx.x;
  if (i < n) p[i] = v;
}

__global__ void k_deg(const long long* __restrict__ row, float* __restrict__ deg, int E) {
  int e = blockIdx.x * blockDim.x + threadIdx.x;
  if (e < E) atomicAdd(&deg[(int)row[e]], 1.0f);
}

__global__ void k_rsqrt(float* d, int n) {
  int i = blockIdx.x * blockDim.x + threadIdx.x;
  if (i < n) d[i] = rsqrtf(d[i]);  // deg >= 1 always (self-loop)
}

// Pre-pack W[K x Ntot] (f32, row-major) into f16 WMMA B-fragment order:
// Wp[(kt*(Ntot/16)+g)*512 + lane*16 + pos] = W[kt*32 + (lane>=16?16:0) + pos][g*16 + (lane&15)]
__global__ void k_pack_w(const float* __restrict__ W, _Float16* __restrict__ Wp,
                         int K, int Ntot) {
  int idx = blockIdx.x * blockDim.x + threadIdx.x;  // over K*Ntot halves
  if (idx >= K * Ntot) return;
  int pos  = idx & 15;
  int lane = (idx >> 4) & 31;
  int blk  = idx >> 9;                 // kt*(Ntot/16) + g
  int g    = blk % (Ntot >> 4);
  int kt   = blk / (Ntot >> 4);
  int n = g * 16 + (lane & 15);
  int k = kt * 32 + ((lane >> 4) << 4) + pos;
  Wp[idx] = (_Float16)W[(size_t)k * Ntot + n];
}

// agg[n][:] = dis[n]^2 * h[n][:]   (self-loop contribution, also initializes agg)
__global__ void k_agg_init(const float* __restrict__ dis, const float* __restrict__ h,
                           float* __restrict__ agg, int n4) {
  int i = blockIdx.x * blockDim.x + threadIdx.x;  // over N * (F_DIM/4) float4s
  if (i >= n4) return;
  int n = i / (F_DIM / 4);
  float s = dis[n] * dis[n];
  float4 v = ((const float4*)h)[i];
  float4 r; r.x = s * v.x; r.y = s * v.y; r.z = s * v.z; r.w = s * v.w;
  ((float4*)agg)[i] = r;
}

// agg[col[e]][:] += dis[row]*dis[col] * h[row[e]][:]   (one wave32 per edge)
__global__ void k_agg_edges(const long long* __restrict__ row, const long long* __restrict__ col,
                            const float* __restrict__ dis, const float* __restrict__ h,
                            float* __restrict__ agg, int E) {
  int e = blockIdx.x * (blockDim.x >> 5) + (threadIdx.x >> 5);
  if (e >= E) return;
  int lane = threadIdx.x & 31;
  int r = (int)row[e], c = (int)col[e];
  float w = dis[r] * dis[c];
  const float* hp = h + (size_t)r * F_DIM + lane * 8;
  float*      ap = agg + (size_t)c * F_DIM + lane * 8;
#pragma unroll
  for (int i = 0; i < 8; ++i) atomicAdd(ap + i, w * hp[i]);
}

// ---------------- WMMA GEMM ----------------
// D[M x Ntot] = (RELU_IN ? relu(A) : A)[M x K] @ W + bias, W pre-packed f16 fragments.
// block: 128 threads = 4 waves, each wave -> 16x64 tile; block tile 64x64.
template <bool RELU_IN>
__global__ __launch_bounds__(128) void k_gemm_wmma(
    const float* __restrict__ A, const _Float16* __restrict__ Wp,
    const float* __restrict__ bias, float* __restrict__ D,
    int M, int K, int Ntot) {
  __shared__ alignas(16) _Float16 ldsB[4 * 32 * 16];  // 4 ntiles x 32 lanes x 16 halves = 4KB

  const int tid  = threadIdx.x;
  const int wave = tid >> 5;
  const int lane = tid & 31;
  const int l16  = lane & 15;
  const int hi   = lane >> 4;

  const int mblk  = blockIdx.x * 64;
  const int ncol0 = blockIdx.y * 64;
  const int wm    = mblk + wave * 16;

  int arow = wm + l16;                  // clamp for tail blocks; stores guarded
  if (arow >= M) arow = M - 1;
  const float* aptr = A + (size_t)arow * K;

  const int nblk = Ntot >> 4;           // fragment blocks per K-tile row
  const int nt0  = ncol0 >> 4;

  v8f acc0 = {}, acc1 = {}, acc2 = {}, acc3 = {};

  for (int k0 = 0; k0 < K; k0 += 32) {
    // ---- stage 4 pre-packed B-fragment blocks (contiguous 4KB) into LDS
    const _Float16* wblk = Wp + ((size_t)(k0 >> 5) * nblk + nt0) * 512;
#if HAVE_ASYNC_LDS
    {
      const char* gsrc = (const char*)wblk + tid * 32;
      char*       ldst = (char*)ldsB + tid * 32;
      __builtin_amdgcn_global_load_async_to_lds_b128(
          (as1_v4i)(uintptr_t)gsrc, (as3_v4i)ldst, 0, 0);
      __builtin_amdgcn_global_load_async_to_lds_b128(
          (as1_v4i)(uintptr_t)(gsrc + 16), (as3_v4i)(ldst + 16), 0, 0);
#  if __has_builtin(__builtin_amdgcn_s_wait_asynccnt)
      __builtin_amdgcn_s_wait_asynccnt(0);
#  else
      asm volatile("s_wait_asynccnt 0" ::: "memory");
#  endif
    }
#else
    {
      const uint4* src = (const uint4*)wblk;
      uint4*       dst = (uint4*)ldsB;
      dst[tid * 2 + 0] = src[tid * 2 + 0];
      dst[tid * 2 + 1] = src[tid * 2 + 1];
    }
#endif
    __syncthreads();

    if (k0 + 32 < K) __builtin_prefetch(aptr + k0 + 32, 0, 1);

    // ---- A fragment: 16x32 f16 (contiguous k = 16*hi + i, same order as packed B)
    v16h afrag;
    {
      const int koff = k0 + (hi ? 16 : 0);
#pragma unroll
      for (int i = 0; i < 8; ++i) {
        float v0 = aptr[koff + i];
        float v1 = aptr[koff + 8 + i];
        if (RELU_IN) { v0 = fmaxf(v0, 0.0f); v1 = fmaxf(v1, 0.0f); }
        afrag[i]     = (_Float16)v0;
        afrag[i + 8] = (_Float16)v1;
      }
    }

    // ---- load all 4 B fragments, then issue 4 WMMAs (batched DS waits)
    v16h b0, b1, b2, b3;
    {
      const _Float16* bp = &ldsB[lane * 16];
#pragma unroll
      for (int i = 0; i < 16; ++i) {
        b0[i] = bp[i];
        b1[i] = bp[512 + i];
        b2[i] = bp[1024 + i];
        b3[i] = bp[1536 + i];
      }
    }
    acc0 = __builtin_amdgcn_wmma_f32_16x16x32_f16(false, afrag, false, b0, (short)0, acc0, false, false);
    acc1 = __builtin_amdgcn_wmma_f32_16x16x32_f16(false, afrag, false, b1, (short)0, acc1, false, false);
    acc2 = __builtin_amdgcn_wmma_f32_16x16x32_f16(false, afrag, false, b2, (short)0, acc2, false, false);
    acc3 = __builtin_amdgcn_wmma_f32_16x16x32_f16(false, afrag, false, b3, (short)0, acc3, false, false);

    __syncthreads();
  }

  // ---- epilogue: bias + store (M is 16-aligned, whole wave in/out of range)
  if (wm < M) {
#pragma unroll
    for (int nt = 0; nt < 4; ++nt) {
      v8f acc = (nt == 0) ? acc0 : (nt == 1) ? acc1 : (nt == 2) ? acc2 : acc3;
      int col = ncol0 + nt * 16 + l16;
      float bv = bias[col];
#pragma unroll
      for (int v = 0; v < 8; ++v) {
        int rrow = wm + v + hi * 8;   // C/D layout: VGPR v -> M = v (+8 for hi lanes)
        D[(size_t)rrow * Ntot + col] = acc[v] + bv;
      }
    }
  }
}

// ---------------- driver ----------------
extern "C" void kernel_launch(void* const* d_in, const int* in_sizes, int n_in,
                              void* d_out, int out_size, void* d_ws, size_t ws_size,
                              hipStream_t stream) {
  const float*     x  = (const float*)d_in[0];
  const long long* ei = (const long long*)d_in[1];   // int64 [2, E]
  const float*     W1 = (const float*)d_in[2];
  const float*     b1 = (const float*)d_in[3];
  const float*     W2 = (const float*)d_in[4];
  const float*     b2 = (const float*)d_in[5];
  const float*     Wh = (const float*)d_in[6];
  const float*     bh = (const float*)d_in[7];
  float*           out = (float*)d_out;

  const int N = in_sizes[0] / F_DIM;
  const int E = in_sizes[1] / 2;
  const long long* row = ei;
  const long long* col = ei + E;

  // workspace: dis[N] | bufA[N*256] | bufB[N*256] | packed f16 weights
  float* dis  = (float*)d_ws;
  float* bufA = dis + (((size_t)N + 255) & ~(size_t)255);
  float* bufB = bufA + (size_t)N * F_DIM;
  _Float16* wp1 = (_Float16*)(bufB + (size_t)N * F_DIM);
  _Float16* wp2 = wp1 + F_DIM * F_DIM;
  _Float16* wph = wp2 + F_DIM * F_DIM;

  // ---- one-time packing of weights into WMMA B-fragment order (f16)
  k_pack_w<<<(F_DIM * F_DIM + 255) / 256, 256, 0, stream>>>(W1, wp1, F_DIM, F_DIM);
  k_pack_w<<<(F_DIM * F_DIM + 255) / 256, 256, 0, stream>>>(W2, wp2, F_DIM, F_DIM);
  k_pack_w<<<(F_DIM * C_DIM + 255) / 256, 256, 0, stream>>>(Wh, wph, F_DIM, C_DIM);

  // ---- sym-norm degrees: deg starts at 1 (self-loop), add over edges, rsqrt
  k_fill  <<<(N + 255) / 256, 256, 0, stream>>>(dis, N, 1.0f);
  k_deg   <<<(E + 255) / 256, 256, 0, stream>>>(row, dis, E);
  k_rsqrt <<<(N + 255) / 256, 256, 0, stream>>>(dis, N);

  const dim3 blk(128);
  const dim3 g1((N + 63) / 64, F_DIM / 64);
  const dim3 g3((N + 63) / 64, C_DIM / 64);
  const int  n4 = N * (F_DIM / 4);
  const int  eb = (E + 7) / 8;   // 8 edges (waves) per 256-thread block

  // layer 1: h = x @ W1 + b1 ; propagate ; (relu fused into next GEMM's A read)
  k_gemm_wmma<false><<<g1, blk, 0, stream>>>(x, wp1, b1, bufA, N, F_DIM, F_DIM);
  k_agg_init <<<(n4 + 255) / 256, 256, 0, stream>>>(dis, bufA, bufB, n4);
  k_agg_edges<<<eb, 256, 0, stream>>>(row, col, dis, bufA, bufB, E);

  // layer 2: h = relu(agg) @ W2 + b2 ; propagate
  k_gemm_wmma<true><<<g1, blk, 0, stream>>>(bufB, wp2, b2, bufA, N, F_DIM, F_DIM);
  k_agg_init <<<(n4 + 255) / 256, 256, 0, stream>>>(dis, bufA, bufB, n4);
  k_agg_edges<<<eb, 256, 0, stream>>>(row, col, dis, bufA, bufB, E);

  // head: out = relu(agg) @ Wh + bh
  k_gemm_wmma<true><<<g3, blk, 0, stream>>>(bufB, wph, bh, out, N, F_DIM, C_DIM);
}
